// Head_64854006170066
// MI455X (gfx1250) — matrix operand order repaired and checked by reading, not compiled, not run on hardware
//
#include <hip/hip_runtime.h>
#include <hip/hip_bf16.h>

typedef __bf16 bf16;
typedef __attribute__((ext_vector_type(16))) __bf16 v16bf;
typedef __attribute__((ext_vector_type(8)))  float  v8f;

constexpr int kT = 256;   // sequence length
constexpr int kC = 384;   // embedding dim
constexpr int kH = 64;    // head size
constexpr int CHUNK = 32; // x rows staged per projection step

// LDS strides padded for bank-conflict-free fragment gathers (64 banks x 4B)
constexpr int XS_STRIDE = 392;  // 384 + 8
constexpr int QK_STRIDE = 72;   // 64 + 8
constexpr int VT_STRIDE = 264;  // 256 + 8

__device__ inline v8f wmma_bf16(v16bf a, v16bf b, v8f c) {
  return __builtin_amdgcn_wmma_f32_16x16x32_bf16(
      /*neg_a=*/false, a, /*neg_b=*/false, b,
      /*c_mod=*/(short)0, c, /*reuse_a=*/false, /*reuse_b=*/false);
}

// A-matrix 16x32 bf16 fragment gather (contiguous K order; B blob uses the same):
// lane L: M = L%16; VGPR jj holds K pair (2jj + 16*(L/16)).
// p points at row M, K-offset 0 of the 32-wide K block.
__device__ inline v16bf a_frag(const bf16* p, int hi) {
  v16bf a;
#pragma unroll
  for (int jj = 0; jj < 8; ++jj) {
    int k = 2 * jj + 16 * hi;
    a[2 * jj]     = p[k];
    a[2 * jj + 1] = p[k + 1];
  }
  return a;
}

// ---------------------------------------------------------------------------
// Kernel 1: gamma = mean(|W|), ternarize, and write combined Wqkv^T [384x192]
// pre-swizzled into WMMA B-fragment order:
//   frag = kb*12 + nt  (kb: K-block of 32 over C, nt: 16-col N-block)
//   element idx = frag*512 + lane*16 + e, value = W_t[c = kb*32+16*(lane/16)+e]
//                                                 [col = nt*16 + lane%16]
// cols 0..63 = Q, 64..127 = K, 128..191 = V.
// ---------------------------------------------------------------------------
__global__ void __launch_bounds__(256)
prep_ternary(const float* __restrict__ Wk, const float* __restrict__ Wq,
             const float* __restrict__ Wv, bf16* __restrict__ blob) {
  __shared__ float red[256];
  __shared__ float gam[3];
  const int tid = threadIdx.x;

  for (int m = 0; m < 3; ++m) {
    const float* W = (m == 0) ? Wq : (m == 1) ? Wk : Wv;
    float s = 0.f;
    for (int i = tid; i < kH * kC; i += 256) s += fabsf(W[i]);
    red[tid] = s;
    __syncthreads();
    for (int off = 128; off > 0; off >>= 1) {
      if (tid < off) red[tid] += red[tid + off];
      __syncthreads();
    }
    if (tid == 0) gam[m] = red[0] / (float)(kH * kC);
    __syncthreads();
  }

  const int total = kC * 192;  // 73728 elements
  for (int idx = tid; idx < total; idx += 256) {
    int frag = idx >> 9;      // /512
    int rem  = idx & 511;
    int L    = rem >> 4;
    int e    = rem & 15;
    int kb = frag / 12, nt = frag % 12;
    int col = nt * 16 + (L & 15);
    int c   = kb * 32 + (L >> 4) * 16 + e;
    int m   = col >> 6;       // 0=Q,1=K,2=V
    int h   = col & 63;
    const float* W = (m == 0) ? Wq : (m == 1) ? Wk : Wv;
    float g = gam[m] + 1e-5f;
    float t = rintf(W[h * kC + c] / g);     // round-half-even, like jnp.round
    t = fminf(1.f, fmaxf(-1.f, t));
    blob[idx] = (bf16)t;
  }
}

// ---------------------------------------------------------------------------
// Kernel 2: fused QKV projection + causal flash attention, one batch / block.
// ---------------------------------------------------------------------------
__global__ void __launch_bounds__(256)
attn_head(const float* __restrict__ x, const bf16* __restrict__ wblob,
          float* __restrict__ out) {
  __shared__ bf16 x_s[CHUNK * XS_STRIDE];   // 25088 B  (bf16 x chunk)
  __shared__ bf16 q_s[kT * QK_STRIDE];      // 36864 B  [t][h]
  __shared__ bf16 k_s[kT * QK_STRIDE];      // 36864 B  [t][h]
  __shared__ bf16 v_t[kH * VT_STRIDE];      // 33792 B  [h][t] (transposed)
  __shared__ bf16 p_s[8 * 16 * 16];         //  4096 B  per-wave P relayout

  const int b   = blockIdx.x;
  const int tid = threadIdx.x;
  const int w   = tid >> 5;     // wave 0..7
  const int l   = tid & 31;
  const int n16 = l & 15;
  const int hi  = l >> 4;

  const float* xb = x + (size_t)b * kT * kC;

  // ---------------- QKV projection: [T x 384] @ [384 x 192] -----------------
  for (int cc = 0; cc < kT / CHUNK; ++cc) {
    const float4* src = (const float4*)(xb + cc * CHUNK * kC);
#pragma unroll
    for (int t = 0; t < (CHUNK * kC / 4) / 256; ++t) {   // 12 float4 / thread
      int f = tid + t * 256;
      int elem = f * 4;
      int row = elem / kC;
      int col = elem % kC;
      float4 v4 = src[f];
      bf16* d = &x_s[row * XS_STRIDE + col];
      d[0] = (bf16)v4.x; d[1] = (bf16)v4.y; d[2] = (bf16)v4.z; d[3] = (bf16)v4.w;
    }
    __syncthreads();

    // 2 M-tiles x 12 N-tiles = 24 output tiles; each wave owns 3.
    for (int tt = w; tt < 24; tt += 8) {
      int mt = tt & 1;
      int nt = tt >> 1;
      v8f acc = {};
#pragma unroll
      for (int kb = 0; kb < kC / 32; ++kb) {
        v16bf a  = a_frag(&x_s[(mt * 16 + n16) * XS_STRIDE + kb * 32], hi);
        v16bf bw = *(const v16bf*)(wblob + (size_t)(kb * 12 + nt) * 512 + l * 16);
        acc = wmma_bf16(a, bw, acc);
      }
      // C-layout: VGPR r -> row (r + 8*hi), col n16
      int rowbase = cc * CHUNK + mt * 16 + 8 * hi;
      if (nt < 4) {
        int h = nt * 16 + n16;
#pragma unroll
        for (int r = 0; r < 8; ++r) q_s[(rowbase + r) * QK_STRIDE + h] = (bf16)acc[r];
      } else if (nt < 8) {
        int h = (nt - 4) * 16 + n16;
#pragma unroll
        for (int r = 0; r < 8; ++r) k_s[(rowbase + r) * QK_STRIDE + h] = (bf16)acc[r];
      } else {
        int h = (nt - 8) * 16 + n16;
#pragma unroll
        for (int r = 0; r < 8; ++r) v_t[h * VT_STRIDE + rowbase + r] = (bf16)acc[r];
      }
    }
    __syncthreads();
  }

  // ---------------- causal flash attention (all operands in LDS) ------------
  const float scale = 0.125f;  // 64^-0.5
  float* outb = out + (size_t)b * kT * kH;

  for (int iq = w; iq < 16; iq += 8) {   // wave owns q-tiles iq and iq+8
    const bf16* qp = &q_s[(iq * 16 + n16) * QK_STRIDE];
    v16bf aq0 = a_frag(qp, hi);          // head dims 0..31
    v16bf aq1 = a_frag(qp + 32, hi);     // head dims 32..63

    v8f accv[4] = {};                    // 16 x 64 output accumulator
    float m_r[8], l_r[8];
#pragma unroll
    for (int r = 0; r < 8; ++r) { m_r[r] = -__builtin_inff(); l_r[r] = 0.f; }

    for (int j = 0; j <= iq; ++j) {
      // B = K_j^T: B[k=headdim][n=keyrow]; lane-half holds headdims 16*hi+e
      v16bf bk0, bk1;
      {
        const bf16* kp = &k_s[(j * 16 + n16) * QK_STRIDE + hi * 16];
#pragma unroll
        for (int e = 0; e < 16; ++e) { bk0[e] = kp[e]; bk1[e] = kp[32 + e]; }
      }
      v8f s = {};
      s = wmma_bf16(aq0, bk0, s);
      s = wmma_bf16(aq1, bk1, s);

      float p_f[8];
      const bool diag = (j == iq);
#pragma unroll
      for (int r = 0; r < 8; ++r) {
        float sv = s[r] * scale;
        if (diag && (n16 > r + 8 * hi)) sv = -__builtin_inff();
        p_f[r] = sv;
      }
      // online softmax: butterfly max/sum across the 16 lanes of each half
#pragma unroll
      for (int r = 0; r < 8; ++r) {
        float mx = p_f[r];
#pragma unroll
        for (int msk = 1; msk < 16; msk <<= 1)
          mx = fmaxf(mx, __shfl_xor(mx, msk, 32));
        float mnew = fmaxf(m_r[r], mx);
        float sf = __expf(m_r[r] - mnew);
        float pe = __expf(p_f[r] - mnew);
        float rs = pe;
#pragma unroll
        for (int msk = 1; msk < 16; msk <<= 1)
          rs += __shfl_xor(rs, msk, 32);
        l_r[r] = l_r[r] * sf + rs;
        m_r[r] = mnew;
        p_f[r] = pe;
#pragma unroll
        for (int tc = 0; tc < 4; ++tc) accv[tc][r] *= sf;
      }

      // Relayout P: C-layout -> LDS -> A-layout (K padded to 32 with zeros)
#pragma unroll
      for (int r = 0; r < 8; ++r)
        p_s[w * 256 + (r + 8 * hi) * 16 + n16] = (bf16)p_f[r];
      asm volatile("s_wait_dscnt 0" ::: "memory");  // same-wave LDS RAW
      v16bf ap = {};
      {
        const bf16* pp = &p_s[w * 256 + n16 * 16];
#pragma unroll
        for (int e = 0; e < 16; ++e) ap[e] = (hi == 0) ? pp[e] : (bf16)0.f;  // K = e (hi==0); hi==1 lanes 0
      }

      // acc += P @ V_j  (only lane-half hi==0 carries valid K rows 0..15)
#pragma unroll
      for (int tc = 0; tc < 4; ++tc) {
        v16bf bv = {};
        if (hi == 0) {
          const bf16* vp = &v_t[(tc * 16 + n16) * VT_STRIDE + j * 16];
#pragma unroll
          for (int e = 0; e < 16; ++e) bv[e] = vp[e];
        }
        accv[tc] = wmma_bf16(ap, bv, accv[tc]);
      }
    }

    // epilogue: out = acc / l
#pragma unroll
    for (int tc = 0; tc < 4; ++tc) {
      int h = tc * 16 + n16;
#pragma unroll
      for (int r = 0; r < 8; ++r) {
        int row = iq * 16 + r + 8 * hi;
        outb[row * kH + h] = accv[tc][r] / l_r[r];
      }
    }
  }
}

extern "C" void kernel_launch(void* const* d_in, const int* in_sizes, int n_in,
                              void* d_out, int out_size, void* d_ws, size_t ws_size,
                              hipStream_t stream) {
  const float* x  = (const float*)d_in[0];
  const float* Wk = (const float*)d_in[1];
  const float* Wq = (const float*)d_in[2];
  const float* Wv = (const float*)d_in[3];
  bf16* wblob = (bf16*)d_ws;              // 384*192 bf16 = 147456 B
  const int B = in_sizes[0] / (kT * kC);  // 512

  prep_ternary<<<1, 256, 0, stream>>>(Wk, Wq, Wv, wblob);
  attn_head<<<B, 256, 0, stream>>>(x, wblob, (float*)d_out);

  (void)n_in; (void)out_size; (void)ws_size;
}
